// LinearAttention_44006234915542
// MI455X (gfx1250) — matrix operand order repaired and checked by reading, not compiled, not run on hardware
//
#include <hip/hip_runtime.h>
#include <hip/hip_bf16.h>
#include <math.h>

typedef __attribute__((ext_vector_type(16))) _Float16 v16h;
typedef __attribute__((ext_vector_type(8)))  float    v8f;

#define DIM      1024
#define HEADS    16
#define HEAD_DIM 64
#define BATCH    4
#define SEQ      4096
#define ROWS     (BATCH * SEQ)      // 16384

// ---------------- CDNA5 async copy (global -> LDS, ASYNCcnt-tracked) -------
__device__ __forceinline__ void async_cp16(unsigned lds_byte, const _Float16* g) {
    asm volatile("global_load_async_to_lds_b128 %0, %1, off"
                 :: "v"(lds_byte), "v"(g) : "memory");
}
#define WAIT_ASYNCCNT(n) asm volatile("s_wait_asynccnt %0" :: "n"(n) : "memory")

// ---------------- fragment helpers (CDNA5 WMMA f16 layouts, ISA 7.12.2) ----
// A 16x32 (f16): lane<16 -> M=lane, K in {kb..kb+7, 16+kb..16+kb+7}, kb=0
//                lane>=16 -> M=lane-16, kb=8
__device__ __forceinline__ v16h load_frag_a(const _Float16* row_base, int stride,
                                            int m, int kb) {
    union { uint4 u[2]; v16h h; } u;
    const _Float16* p = row_base + (size_t)m * stride;
    u.u[0] = *(const uint4*)(p + kb);
    u.u[1] = *(const uint4*)(p + 16 + kb);
    return u.h;
}
// B 32x16 (f16): lane<16 -> N=lane, K in {0..7,16..23} ; lane>=16 -> N=lane-16, K in {8..15,24..31}
// source stored N-major (row n holds K contiguously)
__device__ __forceinline__ v16h load_frag_b(const _Float16* col_base, int stride,
                                            int n, int kbB) {
    union { uint4 u[2]; v16h h; } u;
    const _Float16* p = col_base + (size_t)n * stride;
    u.u[0] = *(const uint4*)(p + (kbB >> 1));
    u.u[1] = *(const uint4*)(p + (kbB >> 1) + 16);
    return u.h;
}

#define WMMA_F16(a, b, c) \
    __builtin_amdgcn_wmma_f32_16x16x32_f16(false, (a), false, (b), (short)0, (c), false, false)

// ---------------- conversion kernels --------------------------------------
__global__ __launch_bounds__(256) void convert_x_kernel(const float* __restrict__ x,
                                                        _Float16* __restrict__ xh) {
    size_t i = ((size_t)blockIdx.x * 256 + threadIdx.x) * 4;
    float4 v = *(const float4*)(x + i);
    union { _Float16 h[4]; uint2 u; } o;
    o.h[0] = (_Float16)v.x; o.h[1] = (_Float16)v.y;
    o.h[2] = (_Float16)v.z; o.h[3] = (_Float16)v.w;
    *(uint2*)(xh + i) = o.u;
}

// W [K][N] f32  ->  Wt [N][K] f16   (32x32 LDS tile transpose)
__global__ __launch_bounds__(1024) void transposeW_kernel(const float* __restrict__ W,
                                                          _Float16* __restrict__ Wt) {
    __shared__ _Float16 t[32][33];
    int tx = threadIdx.x & 31, ty = threadIdx.x >> 5;
    int k = blockIdx.y * 32 + ty, n = blockIdx.x * 32 + tx;
    t[ty][tx] = (_Float16)W[(size_t)k * DIM + n];
    __syncthreads();
    int nn = blockIdx.x * 32 + ty, kk = blockIdx.y * 32 + tx;
    Wt[(size_t)nn * DIM + kk] = t[tx][ty];
}

// ---------------- main 16384x1024x1024 GEMM -------------------------------
// C = act( A[ROWS][1024] * Wt[1024][1024]^T + bias )
// LAYOUT 0: f32 plain row-major     (final projection)
// LAYOUT 1: f16 [b,h,n,d]           (q)
// LAYOUT 2: f16 [b,h,d,n]           (k / v transposed; vector b128 stores)
constexpr int BM = 128, BN = 256, BK = 32;
constexpr int LSTR = BK + 8;   // padded LDS stride (halfs), keeps 16B alignment
constexpr int KSTEPS = DIM / BK;

template <int LAYOUT, bool ACT>
__global__ __launch_bounds__(256) void gemm1024_kernel(
    const _Float16* __restrict__ A, const _Float16* __restrict__ Bt,
    const float* __restrict__ bias, _Float16* __restrict__ outH,
    float* __restrict__ outF) {
    __shared__ alignas(16) _Float16 sA[2][BM * LSTR];
    __shared__ alignas(16) _Float16 sB[2][BN * LSTR];

    const int R0 = blockIdx.x * BM, C0 = blockIdx.y * BN;
    const int tid = threadIdx.x, lane = tid & 31, wid = tid >> 5;
    const int wR = (wid & 1) * 64, wC = (wid >> 1) * 64;  // 2x4 wave grid, 64x64 tiles
    const int m = lane & 15, hi = lane >> 4;
    const int kbA = hi << 3, kbB = hi << 4;

    v8f acc[4][4] = {};

    // async DMA staging of one double-buffer slot: 2 A-chunks + 4 B-chunks/thread
    auto stage = [&](int buf, int k0) {
        unsigned bA = (unsigned)(uintptr_t)&sA[buf][0];
        #pragma unroll
        for (int c = tid; c < 512; c += 256) {          // A: 128 rows x 32 halfs
            int r = c >> 2, ko = (c & 3) << 3;
            async_cp16(bA + (unsigned)(r * LSTR + ko) * 2,
                       A + (size_t)(R0 + r) * DIM + k0 + ko);
        }
        unsigned bB = (unsigned)(uintptr_t)&sB[buf][0];
        #pragma unroll
        for (int c = tid; c < 1024; c += 256) {         // B: 256 rows x 32 halfs
            int r = c >> 2, ko = (c & 3) << 3;
            async_cp16(bB + (unsigned)(r * LSTR + ko) * 2,
                       Bt + (size_t)(C0 + r) * DIM + k0 + ko);
        }
    };

    stage(0, 0);
    for (int it = 0; it < KSTEPS; ++it) {
        const int cur = it & 1;
        if (it + 1 < KSTEPS) {          // overlap next tile's DMA with compute
            stage(cur ^ 1, (it + 1) * BK);
            WAIT_ASYNCCNT(6);           // 6 just issued -> previous 6 complete
        } else {
            WAIT_ASYNCCNT(0);
        }
        __syncthreads();                // all waves' slot `cur` visible

        v16h afr[4], bfr[4];
        #pragma unroll
        for (int i = 0; i < 4; i++) afr[i] = load_frag_a(&sA[cur][(wR + i * 16) * LSTR], LSTR, m, kbA);
        #pragma unroll
        for (int j = 0; j < 4; j++) bfr[j] = load_frag_b(&sB[cur][(wC + j * 16) * LSTR], LSTR, m, kbB);
        #pragma unroll
        for (int i = 0; i < 4; i++)
            #pragma unroll
            for (int j = 0; j < 4; j++)
                acc[i][j] = WMMA_F16(afr[i], bfr[j], acc[i][j]);
        __syncthreads();                // compute done before slot is re-staged
    }

    #pragma unroll
    for (int i = 0; i < 4; i++) {
        #pragma unroll
        for (int j = 0; j < 4; j++) {
            const int Cg = C0 + wC + j * 16 + m;
            const float bv = bias[Cg];
            union { v8f v; float f[8]; } u; u.v = acc[i][j];
            if constexpr (LAYOUT == 2) {
                // e walks rows (=n), contiguous in [b,h,d,n]: one b128 store
                int Rbase = R0 + wR + i * 16 + hi * 8;
                int b = Rbase >> 12, n0 = Rbase & 4095, h = Cg >> 6, d = Cg & 63;
                union { _Float16 h8[8]; uint4 q; } o;
                #pragma unroll
                for (int e = 0; e < 8; e++) {
                    float val = u.f[e] + bv;
                    if (ACT) val = val > 0.f ? val + 1.f : __expf(val);
                    o.h8[e] = (_Float16)val;
                }
                *(uint4*)(outH + ((size_t)(b * HEADS + h) * HEAD_DIM + d) * SEQ + n0) = o.q;
            } else {
                #pragma unroll
                for (int e = 0; e < 8; e++) {
                    int Rg = R0 + wR + i * 16 + (hi ? e + 8 : e);
                    float val = u.f[e] + bv;
                    if (ACT) val = val > 0.f ? val + 1.f : __expf(val);
                    if constexpr (LAYOUT == 0) {
                        outF[(size_t)Rg * DIM + Cg] = val;
                    } else {  // LAYOUT 1: [b,h,n,d]
                        int b = Rg >> 12, n = Rg & 4095, h = Cg >> 6, d = Cg & 63;
                        outH[(((size_t)b * HEADS + h) * SEQ + n) * HEAD_DIM + d] = (_Float16)val;
                    }
                }
            }
        }
    }
}

// ---------------- kv = sum_n k^T v  per (b,h) ------------------------------
// kT, vT: [bh][64][4096] f16 (K-contiguous).  Output kvT: [bh][f][d] f16.
__global__ __launch_bounds__(256) void kv_kernel(const _Float16* __restrict__ kT,
                                                 const _Float16* __restrict__ vT,
                                                 _Float16* __restrict__ kvT) {
    __shared__ float skv[HEAD_DIM * HEAD_DIM];  // [d][f]
    const int bh = blockIdx.x;
    const _Float16* kB = kT + (size_t)bh * HEAD_DIM * SEQ;
    const _Float16* vB = vT + (size_t)bh * HEAD_DIM * SEQ;
    const int tid = threadIdx.x, lane = tid & 31, wid = tid >> 5;
    const int m = lane & 15, hi = lane >> 4;
    const int kbA = hi << 3, kbB = hi << 4;

    for (int i = tid; i < HEAD_DIM * HEAD_DIM; i += 256) skv[i] = 0.f;
    __syncthreads();

    v8f acc[4][4] = {};
    const int nBeg = wid * (SEQ / 8), nEnd = nBeg + (SEQ / 8);
    for (int n0 = nBeg; n0 < nEnd; n0 += 32) {
        v16h afr[4], bfr[4];
        #pragma unroll
        for (int dt = 0; dt < 4; dt++) afr[dt] = load_frag_a(kB + (size_t)dt * 16 * SEQ + n0, SEQ, m, kbA);
        #pragma unroll
        for (int ft = 0; ft < 4; ft++) bfr[ft] = load_frag_b(vB + (size_t)ft * 16 * SEQ + n0, SEQ, m, kbB);
        #pragma unroll
        for (int dt = 0; dt < 4; dt++)
            #pragma unroll
            for (int ft = 0; ft < 4; ft++)
                acc[dt][ft] = WMMA_F16(afr[dt], bfr[ft], acc[dt][ft]);
    }
    #pragma unroll
    for (int dt = 0; dt < 4; dt++)
        #pragma unroll
        for (int ft = 0; ft < 4; ft++) {
            union { v8f v; float f[8]; } u; u.v = acc[dt][ft];
            int f = ft * 16 + m;
            #pragma unroll
            for (int e = 0; e < 8; e++) {
                int d = dt * 16 + (hi ? e + 8 : e);
                atomicAdd(&skv[d * HEAD_DIM + f], u.f[e]);  // ds_add_f32
            }
        }
    __syncthreads();
    for (int i = tid; i < HEAD_DIM * HEAD_DIM; i += 256) {
        int d = i >> 6, f = i & 63;  // store transposed: kvT[f][d]
        kvT[(size_t)bh * HEAD_DIM * HEAD_DIM + f * HEAD_DIM + d] = (_Float16)skv[i];
    }
}

// ---------------- out_h = q @ kv, merge heads ------------------------------
// q: [bh][n][64] f16, kvT: [bh][f][d] f16 -> merged [b][n][1024] f16
__global__ __launch_bounds__(256) void qkv_apply_kernel(const _Float16* __restrict__ q,
                                                        const _Float16* __restrict__ kvT,
                                                        _Float16* __restrict__ merged) {
    const int bh = blockIdx.y, b = bh >> 4, h = bh & 15;
    const _Float16* qB = q + (size_t)bh * SEQ * HEAD_DIM;
    const _Float16* kvB = kvT + (size_t)bh * HEAD_DIM * HEAD_DIM;
    const int tid = threadIdx.x, lane = tid & 31, wid = tid >> 5;
    const int m = lane & 15, hi = lane >> 4;
    const int kbA = hi << 3, kbB = hi << 4;
    const int R0 = blockIdx.x * 256 + wid * 32;

    v8f acc[2][4] = {};
    #pragma unroll
    for (int k0 = 0; k0 < HEAD_DIM; k0 += 32) {
        v16h afr[2], bfr[4];
        #pragma unroll
        for (int i = 0; i < 2; i++) afr[i] = load_frag_a(qB + (size_t)(R0 + i * 16) * HEAD_DIM + k0, HEAD_DIM, m, kbA);
        #pragma unroll
        for (int j = 0; j < 4; j++) bfr[j] = load_frag_b(kvB + (size_t)(j * 16) * HEAD_DIM + k0, HEAD_DIM, m, kbB);
        #pragma unroll
        for (int i = 0; i < 2; i++)
            #pragma unroll
            for (int j = 0; j < 4; j++)
                acc[i][j] = WMMA_F16(afr[i], bfr[j], acc[i][j]);
    }
    #pragma unroll
    for (int i = 0; i < 2; i++)
        #pragma unroll
        for (int j = 0; j < 4; j++) {
            union { v8f v; float f[8]; } u; u.v = acc[i][j];
            int C = h * HEAD_DIM + j * 16 + m;
            #pragma unroll
            for (int e = 0; e < 8; e++) {
                int n = R0 + i * 16 + (hi ? e + 8 : e);
                merged[((size_t)b * SEQ + n) * DIM + C] = (_Float16)u.f[e];
            }
        }
}

// ---------------- host side ------------------------------------------------
extern "C" void kernel_launch(void* const* d_in, const int* in_sizes, int n_in,
                              void* d_out, int out_size, void* d_ws, size_t ws_size,
                              hipStream_t stream) {
    const float* x  = (const float*)d_in[0];
    const float* Wq = (const float*)d_in[1];
    const float* bq = (const float*)d_in[2];
    const float* Wk = (const float*)d_in[3];
    const float* bk = (const float*)d_in[4];
    const float* Wv = (const float*)d_in[5];
    const float* bv = (const float*)d_in[6];
    const float* Wp = (const float*)d_in[7];
    const float* bp = (const float*)d_in[8];
    float* out = (float*)d_out;

    _Float16* ws  = (_Float16*)d_ws;
    _Float16* xh  = ws;                                       // 16384*1024
    _Float16* Wt0 = xh + (size_t)ROWS * DIM;                  // 4 x 1024*1024
    _Float16* Wt1 = Wt0 + (size_t)DIM * DIM;
    _Float16* Wt2 = Wt1 + (size_t)DIM * DIM;
    _Float16* Wt3 = Wt2 + (size_t)DIM * DIM;
    _Float16* qF  = Wt3 + (size_t)DIM * DIM;                  // 16384*1024
    _Float16* kT  = qF + (size_t)ROWS * DIM;
    _Float16* vT  = kT + (size_t)ROWS * DIM;
    _Float16* kvT = vT + (size_t)ROWS * DIM;                  // 64*64*64
    _Float16* merged = xh;  // reuse: xh dead after the three projections

    // 1) precision conversion + weight transposes
    convert_x_kernel<<<(ROWS * DIM) / (256 * 4), 256, 0, stream>>>(x, xh);
    dim3 tgrid(DIM / 32, DIM / 32);
    transposeW_kernel<<<tgrid, 1024, 0, stream>>>(Wq, Wt0);
    transposeW_kernel<<<tgrid, 1024, 0, stream>>>(Wk, Wt1);
    transposeW_kernel<<<tgrid, 1024, 0, stream>>>(Wv, Wt2);
    transposeW_kernel<<<tgrid, 1024, 0, stream>>>(Wp, Wt3);

    // 2) projections (q: elu+1 head-major, k: elu+1 transposed, v: transposed)
    dim3 ggrid(ROWS / BM, DIM / BN);
    gemm1024_kernel<1, true ><<<ggrid, 256, 0, stream>>>(xh, Wt0, bq, qF, nullptr);
    gemm1024_kernel<2, true ><<<ggrid, 256, 0, stream>>>(xh, Wt1, bk, kT, nullptr);
    gemm1024_kernel<2, false><<<ggrid, 256, 0, stream>>>(xh, Wt2, bv, vT, nullptr);

    // 3) kv = sum_n k^T v per (b,h)
    kv_kernel<<<BATCH * HEADS, 256, 0, stream>>>(kT, vT, kvT);

    // 4) out_h = q @ kv, merged heads
    dim3 agrid(SEQ / 256, BATCH * HEADS);
    qkv_apply_kernel<<<agrid, 256, 0, stream>>>(qF, kvT, merged);

    // 5) final projection, f32 output
    gemm1024_kernel<0, false><<<ggrid, 256, 0, stream>>>(merged, Wt3, bp, nullptr, out);
}
